// MoEMultiscaleINR_89635967468089
// MI455X (gfx1250) — matrix operand order repaired and checked by reading, not compiled, not run on hardware
//
#include <hip/hip_runtime.h>
#include <hip/hip_bf16.h>

#define TILE_M   64
#define NTHREADS 512
#define NGEMM    80
#define NEXP     6

// padded LDS strides (halves) to avoid bank conflicts on ds_load_b128
#define WSTRIDE  520   // work buffers A/B/C (row = 1040B, 16B aligned, rotates 4 banks/row)
#define XSTRIDE  40    // enc buffer

typedef __attribute__((ext_vector_type(16))) _Float16 v16h;
typedef __attribute__((ext_vector_type(8)))  float    v8f;

struct EOp { unsigned char type, src, dst, comb; unsigned short K, F; float omega; };
// type: 0=gemm+sine, 1=gemm+relu, 2=combine dst=0.5*(dst+comb) width F,
//       3=copy A[:,256:512]=C[:,0:256], 4=accumulate out += w[e]*C[:,0:3]
// buffers: 0=XE(enc, stride XSTRIDE), 1=A, 2=B, 3=C (stride WSTRIDE)

#define STRUCT_LIST \
  {0,0,2,0,26,256,30.f},{0,2,1,0,256,256,30.f},{0,0,2,0,26,256,30.f},{2,0,1,2,0,256,0.f}, \
  {0,1,3,0,256,512,30.f},{0,3,2,0,512,512,30.f},{0,1,3,0,256,512,30.f},{2,0,2,3,0,512,0.f}, \
  {0,2,3,0,512,512,30.f},{0,3,1,0,512,512,30.f},{2,0,1,2,0,512,0.f}, \
  {0,1,3,0,512,256,30.f},{0,3,2,0,256,256,30.f},{0,1,3,0,512,256,30.f},{2,0,2,3,0,256,0.f}, \
  {0,2,3,0,256,512,30.f},{0,3,1,0,512,512,30.f},{0,2,3,0,256,512,30.f},{2,0,1,3,0,512,0.f}, \
  {0,1,3,0,512,256,30.f},{0,3,2,0,256,256,30.f},{0,1,3,0,512,256,30.f},{2,0,2,3,0,256,0.f}, \
  {0,2,3,0,256,256,30.f},{0,3,1,0,256,256,30.f},{2,0,1,2,0,256,0.f}, \
  {0,1,3,0,256,3,20.f},{4,0,0,0,0,0,0.f}
#define N_STRUCT 28

#define DETAIL_LIST \
  {0,0,1,0,26,256,50.f},{0,1,2,0,256,512,60.f},{0,2,1,0,512,256,60.f}, \
  {0,1,2,0,256,512,70.f},{0,2,1,0,512,256,70.f},{0,1,3,0,256,3,50.f},{4,0,0,0,0,0,0.f}
#define N_DETAIL 7

#define HYBRID_LIST \
  {0,0,2,0,26,256,30.f},{0,2,1,0,256,256,30.f},{0,0,2,0,26,256,30.f},{2,0,1,2,0,256,0.f}, \
  {0,1,3,0,256,512,30.f},{0,3,2,0,512,512,30.f},{0,1,3,0,256,512,30.f},{2,0,2,3,0,512,0.f}, \
  {0,2,3,0,512,256,30.f},{0,3,1,0,256,256,30.f},{0,2,3,0,512,256,30.f},{2,0,1,3,0,256,0.f}, \
  {0,0,3,0,26,256,50.f},{0,3,2,0,256,512,60.f},{0,2,3,0,512,256,60.f}, \
  {3,0,0,0,0,0,0.f},{1,1,2,0,512,256,0.f},{0,2,3,0,256,3,30.f},{4,0,0,0,0,0,0.f}
#define N_HYBRID 19

__device__ static const EOp d_struct_ops[N_STRUCT] = { STRUCT_LIST };
__device__ static const EOp d_detail_ops[N_DETAIL] = { DETAIL_LIST };
__device__ static const EOp d_hybrid_ops[N_HYBRID] = { HYBRID_LIST };

static const EOp h_struct_ops[N_STRUCT] = { STRUCT_LIST };
static const EOp h_detail_ops[N_DETAIL] = { DETAIL_LIST };
static const EOp h_hybrid_ops[N_HYBRID] = { HYBRID_LIST };

// ---------------- pack kernel: fp32 weights -> f16 WMMA B-fragment layout ----------------
// tile (kt,nt) of 32x16; element index = tile*512 + lane*16 + h
// lane<16: K = kt*32 + (h<8 ? h : 16+h-8),   N = lane
// lane>=16: K = kt*32 + (h<8 ? 8+h : 24+h-8), N = lane-16
struct PackArgs {
  const float* src[NGEMM];
  unsigned long long off[NGEMM];     // in halves
  unsigned short K[NGEMM], F[NGEMM];
};

__global__ __launch_bounds__(256) void pack_kernel(PackArgs pa, _Float16* __restrict__ wpk) {
  const int mi = blockIdx.y;
  const unsigned K = pa.K[mi], F = pa.F[mi];
  const unsigned Kp = (K + 31u) & ~31u, Fp = (F + 15u) & ~15u;
  const unsigned KT = Kp >> 5;
  const unsigned total = KT * (Fp >> 4) * 512u;
  const unsigned i = blockIdx.x * 256u + threadIdx.x;
  if (i >= total) return;
  const unsigned tile = i >> 9, rem = i & 511u;
  const unsigned lane = rem >> 4, h = rem & 15u;
  const unsigned nt = tile / KT, kt = tile % KT;
  const unsigned k = kt * 32u + ((h & 8u) ? 16u : 0u) + (lane >> 4) * 8u + (h & 7u);
  const unsigned col = nt * 16u + (lane & 15u);
  float v = (k < K && col < F) ? pa.src[mi][(size_t)k * F + col] : 0.f;
  wpk[pa.off[mi] + i] = (_Float16)v;
}

// ---------------- main fused MoE kernel ----------------
struct MainArgs {
  unsigned long long woff[NGEMM];    // packed weight offsets (halves)
  const float* bias[NGEMM];
  const float* coords;
  const float* complexity;
  const float* r1w; const float* r1b; const float* r2w; const float* r2b;
  float* out;
  const _Float16* wpk;
};

// LDS layout (halves): XE @0 (64x40), A @2560, B @35840, C @69120 (each 64x520)
// WGT(f32) @byte 204800 (64x6), OUT(f32) @byte 206336 (64x3). Total 207104 B.
#define SMEM_BYTES 207104

__device__ __forceinline__ _Float16* bufp(char* smem, int id) {
  int off = (id == 0) ? 0 : (id == 1) ? 2560 : (id == 2) ? 35840 : 69120;
  return (_Float16*)smem + off;
}

__device__ __forceinline__ void gemm_layer(
    const _Float16* __restrict__ wmat, const float* __restrict__ bias,
    const _Float16* __restrict__ src, int sstride, _Float16* __restrict__ dst,
    int K, int F, float omega, int isrelu)
{
  const int Kp = (K + 31) & ~31, Fp = (F + 15) & ~15;
  const int TM = TILE_M / 16, TN = Fp >> 4, KT = Kp >> 5;
  const int jobs = TM * TN;
  const int wave = threadIdx.x >> 5, lane = threadIdx.x & 31;
  const int nwaves = NTHREADS / 32;
  const int grp = lane >> 4, n = lane & 15;
  for (int job = wave; job < jobs; job += nwaves) {
    const int mt = job % TM, nt = job / TM;
    v8f acc = {};
    // A fragment (16x32 f16): row = mt*16 + (lane%16); halves 0..7 at K+grp*8, 8..15 at +16
    const _Float16* ap = src + (mt * 16 + n) * sstride + grp * 8;
    // B fragment: contiguous 32B per lane in packed layout
    const _Float16* bp = wmat + (size_t)(nt * KT) * 512 + lane * 16;
    for (int kt = 0; kt < KT; kt++) {
      union { v16h v; uint4 q[2]; } af, bf;
      af.q[0] = *(const uint4*)(ap);
      af.q[1] = *(const uint4*)(ap + 16);
      ap += 32;
      bf.q[0] = *(const uint4*)(bp);
      bf.q[1] = *(const uint4*)(bp + 8);
      bp += 512;
      acc = __builtin_amdgcn_wmma_f32_16x16x32_f16(false, af.v, false, bf.v,
                                                   (short)0, acc, false, false);
    }
    const int col = nt * 16 + n;
    const float bj = (col < F) ? bias[col] : 0.f;
#pragma unroll
    for (int r = 0; r < 8; r++) {
      const int m = mt * 16 + grp * 8 + r;   // C layout: VGPR r -> M = r (+8 for upper lanes)
      float v = acc[r] + bj;
      v = isrelu ? fmaxf(v, 0.f) : __sinf(omega * v);
      dst[m * WSTRIDE + col] = (_Float16)v;
    }
  }
}

__global__ __launch_bounds__(NTHREADS) void moe_kernel(MainArgs args) {
  extern __shared__ __align__(16) char smem[];
  _Float16* XE = (_Float16*)smem;
  float* WGT  = (float*)(smem + 204800);
  float* OUTB = (float*)(smem + 206336);
  const int tid = threadIdx.x;
  const int tile0 = blockIdx.x * TILE_M;
  const float PI = 3.14159265358979323846f;

  // positional encoding -> XE (f16, cols 0..25 valid, 26..31 zero; cols 32..39 never read)
  for (int i = tid; i < TILE_M * 32; i += NTHREADS) {
    const int m = i >> 5, c = i & 31;
    const float x0 = args.coords[(size_t)(tile0 + m) * 2];
    const float x1 = args.coords[(size_t)(tile0 + m) * 2 + 1];
    float v;
    if (c == 0) v = x0;
    else if (c == 1) v = x1;
    else if (c < 26) {
      const int t = c - 2, fi = t >> 2, s = (t >> 1) & 1, d = t & 1;
      const float xf = (d ? x1 : x0) * PI * (float)(1 << fi);
      v = s ? __cosf(xf) : __sinf(xf);
    } else v = 0.f;
    XE[m * XSTRIDE + c] = (_Float16)v;
  }
  for (int i = tid; i < TILE_M * 3; i += NTHREADS) OUTB[i] = 0.f;
  __syncthreads();

  // router (one point per thread, f32)
  if (tid < TILE_M) {
    const int gm = tile0 + tid;
    const float x0 = args.coords[(size_t)gm * 2], x1 = args.coords[(size_t)gm * 2 + 1];
    float enc[26];
    enc[0] = x0; enc[1] = x1;
#pragma unroll
    for (int i = 0; i < 6; i++) {
      const float f = PI * (float)(1 << i);
      enc[2 + i * 4 + 0] = __sinf(x0 * f);
      enc[2 + i * 4 + 1] = __sinf(x1 * f);
      enc[2 + i * 4 + 2] = __cosf(x0 * f);
      enc[2 + i * 4 + 3] = __cosf(x1 * f);
    }
    float logit[6];
#pragma unroll
    for (int e = 0; e < 6; e++) logit[e] = args.r2b[e];
    for (int j = 0; j < 128; j++) {
      float h = args.r1b[j];
      for (int c = 0; c < 26; c++) h += enc[c] * args.r1w[c * 128 + j];
      h = fmaxf(h, 0.f);
#pragma unroll
      for (int e = 0; e < 6; e++) logit[e] += h * args.r2w[j * 6 + e];
    }
    const float comp = args.complexity[gm];
    const float sf[6] = {2.f, 2.f, 1.f, 1.f, .5f, .5f};
    const float mf[6] = {.5f, .5f, 2.f, 2.f, .5f, .5f};
    const float cf[6] = {.5f, .5f, 1.f, 1.f, 2.f, 2.f};
    float z[6], mx = -1e30f;
#pragma unroll
    for (int e = 0; e < 6; e++) {
      const float fac = (comp <= 0.01f) ? sf[e] : ((comp <= 0.01f) ? mf[e] : cf[e]);
      z[e] = logit[e] * fac * 10.f;   // / TEMP(0.1)
      mx = fmaxf(mx, z[e]);
    }
    float sum = 0.f;
#pragma unroll
    for (int e = 0; e < 6; e++) { z[e] = __expf(z[e] - mx); sum += z[e]; }
    const float inv = 1.f / sum;
#pragma unroll
    for (int e = 0; e < 6; e++) WGT[tid * 6 + e] = z[e] * inv;
  }
  __syncthreads();

  // interpret expert programs
  int slot = 0;
  for (int e = 0; e < NEXP; e++) {
    const EOp* prog = (e < 2) ? d_struct_ops : (e < 4) ? d_detail_ops : d_hybrid_ops;
    const int nops  = (e < 2) ? N_STRUCT     : (e < 4) ? N_DETAIL     : N_HYBRID;
    for (int o = 0; o < nops; o++) {
      const EOp op = prog[o];
      if (op.type <= 1) {
        gemm_layer(args.wpk + args.woff[slot], args.bias[slot],
                   bufp(smem, op.src), (op.src == 0) ? XSTRIDE : WSTRIDE,
                   bufp(smem, op.dst), op.K, op.F, op.omega, op.type == 1);
        slot++;
      } else if (op.type == 2) {
        const int W = op.F, sh = (W == 512) ? 9 : 8;
        _Float16* d = bufp(smem, op.dst);
        const _Float16* c = bufp(smem, op.comb);
        for (int i = tid; i < TILE_M * W; i += NTHREADS) {
          const int m = i >> sh, cc = i & (W - 1);
          const int idx = m * WSTRIDE + cc;
          d[idx] = (_Float16)(0.5f * ((float)d[idx] + (float)c[idx]));
        }
      } else if (op.type == 3) {
        _Float16* a = bufp(smem, 1);
        const _Float16* c = bufp(smem, 3);
        for (int i = tid; i < TILE_M * 256; i += NTHREADS) {
          const int m = i >> 8, cc = i & 255;
          a[m * WSTRIDE + 256 + cc] = c[m * WSTRIDE + cc];
        }
      } else {
        const _Float16* c = bufp(smem, 3);
        for (int i = tid; i < TILE_M * 3; i += NTHREADS) {
          const int m = i / 3, j = i - m * 3;
          OUTB[i] += WGT[m * 6 + e] * (float)c[m * WSTRIDE + j];
        }
      }
      __syncthreads();
    }
  }

  for (int i = tid; i < TILE_M * 3; i += NTHREADS)
    args.out[(size_t)tile0 * 3 + i] = OUTB[i];
}

// ---------------- host launcher ----------------
extern "C" void kernel_launch(void* const* d_in, const int* in_sizes, int n_in,
                              void* d_out, int out_size, void* d_ws, size_t ws_size,
                              hipStream_t stream) {
  (void)n_in; (void)out_size; (void)ws_size;
  PackArgs pa;
  MainArgs ma;
  ma.coords     = (const float*)d_in[0];
  ma.complexity = (const float*)d_in[1];
  ma.r1w = (const float*)d_in[2];
  ma.r1b = (const float*)d_in[3];
  ma.r2w = (const float*)d_in[4];
  ma.r2b = (const float*)d_in[5];
  ma.out = (float*)d_out;
  ma.wpk = (const _Float16*)d_ws;

  int di = 6, slot = 0;
  unsigned long long off = 0;
  for (int e = 0; e < NEXP; e++) {
    const EOp* prog = (e < 2) ? h_struct_ops : (e < 4) ? h_detail_ops : h_hybrid_ops;
    const int nops  = (e < 2) ? N_STRUCT     : (e < 4) ? N_DETAIL     : N_HYBRID;
    for (int o = 0; o < nops; o++) {
      const EOp op = prog[o];
      if (op.type <= 1) {
        const unsigned K = op.K, F = op.F;
        const unsigned Kp = (K + 31u) & ~31u, Fp = (F + 15u) & ~15u;
        pa.src[slot] = (const float*)d_in[di];
        ma.bias[slot] = (const float*)d_in[di + 1];
        di += 2;
        pa.K[slot] = (unsigned short)K;
        pa.F[slot] = (unsigned short)F;
        pa.off[slot] = off;
        ma.woff[slot] = off;
        off += (unsigned long long)(Kp >> 5) * (Fp >> 4) * 512ull;
        slot++;
      }
    }
  }

  // pack all 80 weight matrices into f16 fragment layout in d_ws (~17.3 MB)
  pack_kernel<<<dim3(1024, NGEMM, 1), 256, 0, stream>>>(pa, (_Float16*)d_ws);

  const int N = in_sizes[0] / 2;
  moe_kernel<<<dim3(N / TILE_M, 1, 1), NTHREADS, SMEM_BYTES, stream>>>(ma);
}
